// GraphSAGE_24610162606526
// MI455X (gfx1250) — compile-verified, matrix operand order checked by reading, not run on hardware
//
#include <hip/hip_runtime.h>
#include <hip/hip_bf16.h>

#define N_NODES 50000
#define DIM     256

typedef __attribute__((ext_vector_type(16))) __bf16 v16bf;
typedef __attribute__((ext_vector_type(8)))  __bf16 v8bf;
typedef __attribute__((ext_vector_type(8)))  float  v8f;

// Build a 16x32-bf16 WMMA operand register file for one lane from two 16-byte
// chunks. Per CDNA5 ISA layout, lanes 0-15 hold K={0..7,16..23}, lanes 16-31
// hold K={8..15,24..31}; caller passes the right base offsets.
__device__ __forceinline__ v16bf load_ab(const __bf16* p0, const __bf16* p1) {
    v8bf lo = *reinterpret_cast<const v8bf*>(p0);
    v8bf hi = *reinterpret_cast<const v8bf*>(p1);
    v16bf r;
#pragma unroll
    for (int i = 0; i < 8; ++i) { r[i] = lo[i]; r[i + 8] = hi[i]; }
    return r;
}

__global__ void zero_f32_kernel(float* __restrict__ p, long n) {
    long i = (long)blockIdx.x * blockDim.x + threadIdx.x;
    long stride = (long)gridDim.x * blockDim.x;
    for (; i < n; i += stride) p[i] = 0.0f;
}

__global__ void degree_kernel(const int* __restrict__ tgt, float* __restrict__ cnt, int E) {
    int i = blockIdx.x * blockDim.x + threadIdx.x;
    if (i < E) atomicAdd(&cnt[tgt[i]], 1.0f);
}

// scatter-add of fp32 node features: each thread moves 4 floats of one edge
__global__ void scatter_f32_kernel(const float4* __restrict__ x4,
                                   const int* __restrict__ src,
                                   const int* __restrict__ tgt,
                                   float* __restrict__ sum, int E) {
    int i = blockIdx.x * blockDim.x + threadIdx.x;
    int total = E * 64;                    // 64 float4 groups per edge row
    if (i >= total) return;
    int e = i >> 6, q = i & 63;
    float4 v = x4[(long)src[e] * (DIM / 4) + q];
    float* d = sum + (long)tgt[e] * DIM + q * 4;
    atomicAdd(d + 0, v.x);
    atomicAdd(d + 1, v.y);
    atomicAdd(d + 2, v.z);
    atomicAdd(d + 3, v.w);
}

// scatter-add reading bf16 hidden features, accumulating fp32
__global__ void scatter_bf16_kernel(const __bf16* __restrict__ h,
                                    const int* __restrict__ src,
                                    const int* __restrict__ tgt,
                                    float* __restrict__ sum, int E) {
    int i = blockIdx.x * blockDim.x + threadIdx.x;
    int total = E * 64;
    if (i >= total) return;
    int e = i >> 6, q = i & 63;
    const __bf16* s = h + (long)src[e] * DIM + q * 4;
    float* d = sum + (long)tgt[e] * DIM + q * 4;
    atomicAdd(d + 0, (float)s[0]);
    atomicAdd(d + 1, (float)s[1]);
    atomicAdd(d + 2, (float)s[2]);
    atomicAdd(d + 3, (float)s[3]);
}

// agg = sum / max(cnt,1) -> bf16 ; optionally also convert x -> bf16
__global__ void mean_cvt_kernel(const float* __restrict__ sum,
                                const float* __restrict__ cnt,
                                __bf16* __restrict__ aggb,
                                const float* __restrict__ x,
                                __bf16* __restrict__ xb, long n) {
    long i = (long)blockIdx.x * blockDim.x + threadIdx.x;
    long stride = (long)gridDim.x * blockDim.x;
    for (; i < n; i += stride) {
        float c = cnt[i >> 8];
        c = c > 1.0f ? c : 1.0f;
        aggb[i] = (__bf16)(sum[i] / c);
        if (xb) xb[i] = (__bf16)x[i];
    }
}

// Wt[n][k] = bf16(W[k][n]) : column-major bf16 copy so WMMA B-operand
// (column n, contiguous K) loads with the same per-lane pattern as A.
__global__ void wt_kernel(const float* __restrict__ W, __bf16* __restrict__ Wt) {
    int i = blockIdx.x * blockDim.x + threadIdx.x;   // DIM*DIM threads
    if (i >= DIM * DIM) return;
    int k = i >> 8, n = i & 255;
    Wt[n * DIM + k] = (__bf16)W[i];
}

// One wave computes one 16x16 output tile of: Aagg @ Wl + Aself @ Wr (+bias,[relu])
// K = 256 -> 8 unrolled steps x 2 WMMAs = 16 v_wmma_f32_16x16x32_bf16 per wave.
__global__ __launch_bounds__(128)
void sage_gemm_kernel(const __bf16* __restrict__ Aagg,
                      const __bf16* __restrict__ Aself,
                      const __bf16* __restrict__ BlT,
                      const __bf16* __restrict__ BrT,
                      const float*  __restrict__ bias,
                      float* __restrict__ outF,
                      __bf16* __restrict__ outB,
                      int relu, int nTiles) {
    int lane = threadIdx.x & 31;
    int tile = blockIdx.x * (blockDim.x >> 5) + (threadIdx.x >> 5);
    if (tile >= nTiles) return;                 // wave-uniform; EXEC stays all-ones
    int m0 = (tile >> 4) << 4;                  // node-tile base row
    int n0 = (tile & 15) << 4;                  // feature-tile base col
    int lr = lane & 15;
    int lh = (lane >> 4) << 3;                  // 0 for lanes 0-15, 8 for 16-31

    const __bf16* a1 = Aagg  + (long)(m0 + lr) * DIM;
    const __bf16* a2 = Aself + (long)(m0 + lr) * DIM;
    const __bf16* b1 = BlT   + (long)(n0 + lr) * DIM;
    const __bf16* b2 = BrT   + (long)(n0 + lr) * DIM;

    v8f c = {};
#pragma unroll
    for (int kb = 0; kb < DIM; kb += 32) {
        v16bf va1 = load_ab(a1 + kb + lh, a1 + kb + 16 + lh);
        v16bf vb1 = load_ab(b1 + kb + lh, b1 + kb + 16 + lh);
        c = __builtin_amdgcn_wmma_f32_16x16x32_bf16(false, va1, false, vb1,
                                                    (short)0, c, false, false);
        v16bf va2 = load_ab(a2 + kb + lh, a2 + kb + 16 + lh);
        v16bf vb2 = load_ab(b2 + kb + lh, b2 + kb + 16 + lh);
        c = __builtin_amdgcn_wmma_f32_16x16x32_bf16(false, va2, false, vb2,
                                                    (short)0, c, false, false);
    }

    float bb  = bias[n0 + lr];
    int   col = n0 + lr;
#pragma unroll
    for (int r = 0; r < 8; ++r) {
        float v = c[r] + bb;
        if (relu) v = v > 0.0f ? v : 0.0f;
        long o = (long)(m0 + lh + r) * DIM + col;   // C/D layout: VGPR r -> row lh+r
        if (outF) outF[o] = v;
        if (outB) outB[o] = (__bf16)v;
    }
}

extern "C" void kernel_launch(void* const* d_in, const int* in_sizes, int n_in,
                              void* d_out, int out_size, void* d_ws, size_t ws_size,
                              hipStream_t stream) {
    const float* x   = (const float*)d_in[0];
    const int*   ei  = (const int*)d_in[1];
    const float* W1l = (const float*)d_in[2];
    const float* b1  = (const float*)d_in[3];
    const float* W1r = (const float*)d_in[4];
    const float* W2l = (const float*)d_in[5];
    const float* b2  = (const float*)d_in[6];
    const float* W2r = (const float*)d_in[7];

    const int E   = in_sizes[1] / 2;
    const int* src = ei;
    const int* tgt = ei + E;

    const long ND = (long)N_NODES * DIM;
    const int  nTiles = (N_NODES / 16) * (DIM / 16);   // 50000 (exact)

    // Workspace carve-out (256B aligned slices)
    char* ws = (char*)d_ws;
    size_t off = 0;
    auto take = [&](size_t bytes) -> void* {
        void* p = ws + off;
        off += (bytes + 255) & ~(size_t)255;
        return p;
    };
    float*  sum  = (float*) take(ND * 4);
    float*  cnt  = (float*) take((size_t)N_NODES * 4);
    __bf16* xb   = (__bf16*)take(ND * 2);
    __bf16* aggb = (__bf16*)take(ND * 2);
    __bf16* hb   = (__bf16*)take(ND * 2);
    __bf16* Wt1l = (__bf16*)take((size_t)DIM * DIM * 2);
    __bf16* Wt1r = (__bf16*)take((size_t)DIM * DIM * 2);
    __bf16* Wt2l = (__bf16*)take((size_t)DIM * DIM * 2);
    __bf16* Wt2r = (__bf16*)take((size_t)DIM * DIM * 2);

    const int scatterTotal  = E * 64;
    const int scatterBlocks = (scatterTotal + 255) / 256;
    const int edgeBlocks    = (E + 255) / 256;
    const int wtBlocks      = (DIM * DIM + 255) / 256;
    const int gemmBlocks    = (nTiles + 3) / 4;        // 4 waves / block

    // ---- Layer 1 ----
    zero_f32_kernel<<<4096, 256, 0, stream>>>(sum, ND);
    zero_f32_kernel<<<256,  256, 0, stream>>>(cnt, N_NODES);
    degree_kernel<<<edgeBlocks, 256, 0, stream>>>(tgt, cnt, E);
    scatter_f32_kernel<<<scatterBlocks, 256, 0, stream>>>((const float4*)x, src, tgt, sum, E);
    mean_cvt_kernel<<<8192, 256, 0, stream>>>(sum, cnt, aggb, x, xb, ND);

    wt_kernel<<<wtBlocks, 256, 0, stream>>>(W1l, Wt1l);
    wt_kernel<<<wtBlocks, 256, 0, stream>>>(W1r, Wt1r);
    wt_kernel<<<wtBlocks, 256, 0, stream>>>(W2l, Wt2l);
    wt_kernel<<<wtBlocks, 256, 0, stream>>>(W2r, Wt2r);

    sage_gemm_kernel<<<gemmBlocks, 128, 0, stream>>>(aggb, xb, Wt1l, Wt1r, b1,
                                                     nullptr, hb, /*relu=*/1, nTiles);

    // ---- Layer 2 ----
    zero_f32_kernel<<<4096, 256, 0, stream>>>(sum, ND);
    scatter_bf16_kernel<<<scatterBlocks, 256, 0, stream>>>(hb, src, tgt, sum, E);
    mean_cvt_kernel<<<8192, 256, 0, stream>>>(sum, cnt, aggb, nullptr, nullptr, ND);

    sage_gemm_kernel<<<gemmBlocks, 128, 0, stream>>>(aggb, hb, Wt2l, Wt2r, b2,
                                                     (float*)d_out, nullptr, /*relu=*/0, nTiles);
}
